// JobActorCriticAgent_74242804679193
// MI455X (gfx1250) — matrix operand order repaired and checked by reading, not compiled, not run on hardware
//
#include <hip/hip_runtime.h>
#include <hip/hip_bf16.h>
#include <math.h>

#define NNODES 524288
#define NEDGES 4194304
#define INDIM 6
#define HID 64
#define NENVS 64
#define NJOBS 64
#define NGRAPHS (NENVS * NJOBS)             // 4096
#define NODES_PER_GRAPH (NNODES / NGRAPHS)  // 128

typedef __attribute__((ext_vector_type(16))) _Float16 v16h;
typedef __attribute__((ext_vector_type(8)))  float    v8f;

// ---------------------------------------------------------------------------
// WMMA fragment helpers (wave32, A/B k order used by this code, D per CDNA5 ISA 7.12.2)
//
//  A (16-bit, 16x32): lane<16 -> row=lane,   elem j = K j
//                     lane>=16 -> row=lane-16, elem j = K 16+j
//  B (16-bit, 32x16): lane<16 -> col=lane,   elem j = K j
//                     lane>=16 -> col=lane-16, elem j = K 16+j
//  D (f32, 16x16): vgpr r: lane<16 -> (row r, col lane); lane>=16 -> (row 8+r, col lane-16)
//
// W (64x64 f32, row-major W[k][c]) is swizzled into LDS in B-fragment order:
//   Wswz[ ((kh*4+n)*2+sel)*256 + col*16 + j ] = W[kh*32+sel*16+j][n*16+col]
// so each lane reads its 16 B elements as one 32-byte contiguous, 32B-aligned
// chunk (2x ds_load_b128) instead of 16x ds_load_u16.
// ---------------------------------------------------------------------------

__device__ __forceinline__ void fill_wswz(_Float16* __restrict__ Wswz,
                                          const float* __restrict__ W,
                                          int t, int tstride) {
    for (int i = t; i < 8192; i += tstride) {
        int j   = i & 15;
        int c   = (i >> 4) & 15;
        int blk = i >> 8;             // (kh*4+n)*2+sel
        int sel = blk & 1;
        int n   = (blk >> 1) & 3;
        int kh  = blk >> 3;
        Wswz[i] = (_Float16)W[(kh * 32 + sel * 16 + j) * HID + n * 16 + c];
    }
}

__device__ __forceinline__ void load_bfrags(const _Float16* __restrict__ Wswz,
                                            v16h bf[8], int sel, int col) {
#pragma unroll
    for (int kb = 0; kb < 8; ++kb)   // kb = kh*4+n
        bf[kb] = *reinterpret_cast<const v16h*>(Wswz + ((kb * 2 + sel) * 16 + col) * 16);
}

__device__ __forceinline__ v16h make_afrag(const float* __restrict__ Arows,
                                           int row, int sel, int kbase) {
    const float4* Ar = reinterpret_cast<const float4*>(Arows + row * HID + kbase);
    float4 p0 = Ar[sel * 4 + 0];      // K = kbase + sel*16 + 0..3
    float4 p1 = Ar[sel * 4 + 1];      // K = kbase + sel*16 + 4..7
    float4 p2 = Ar[sel * 4 + 2];      // K = kbase + sel*16 + 8..11
    float4 p3 = Ar[sel * 4 + 3];      // K = kbase + sel*16 + 12..15
    v16h a;
    a[0]  = (_Float16)p0.x;  a[1]  = (_Float16)p0.y;
    a[2]  = (_Float16)p0.z;  a[3]  = (_Float16)p0.w;
    a[4]  = (_Float16)p1.x;  a[5]  = (_Float16)p1.y;
    a[6]  = (_Float16)p1.z;  a[7]  = (_Float16)p1.w;
    a[8]  = (_Float16)p2.x;  a[9]  = (_Float16)p2.y;
    a[10] = (_Float16)p2.z;  a[11] = (_Float16)p2.w;
    a[12] = (_Float16)p3.x;  a[13] = (_Float16)p3.y;
    a[14] = (_Float16)p3.z;  a[15] = (_Float16)p3.w;
    return a;
}

// One 16x64 output tile, K=64, B fragments pinned in registers.
__device__ __forceinline__ void wave_gemm_tile(const float* __restrict__ Arows,
                                               const v16h bf[8], v8f acc[4],
                                               int row, int sel) {
#pragma unroll
    for (int kh = 0; kh < 2; ++kh) {
        v16h a = make_afrag(Arows, row, sel, kh * 32);
#pragma unroll
        for (int n = 0; n < 4; ++n)
            acc[n] = __builtin_amdgcn_wmma_f32_16x16x32_f16(
                false, a, false, bf[kh * 4 + n], (short)0, acc[n], false, false);
    }
}

// ---------------------------------------------------------------------------
// Degree / normalization
// ---------------------------------------------------------------------------
__global__ void k_init_deg(float* __restrict__ deg) {
    int i = blockIdx.x * blockDim.x + threadIdx.x;
    if (i < NNODES) deg[i] = 1.0f;  // self-loop contributes 1
}

__global__ void k_deg_accum(const int* __restrict__ dst, float* __restrict__ deg) {
    int e = blockIdx.x * blockDim.x + threadIdx.x;
    if (e < NEDGES) atomicAdd(&deg[dst[e]], 1.0f);
}

__global__ void k_rsqrt_inplace(float* __restrict__ d) {
    int i = blockIdx.x * blockDim.x + threadIdx.x;
    if (i < NNODES) d[i] = rsqrtf(d[i]);
}

// ---------------------------------------------------------------------------
// GCN layer 1 transform: h = features @ W1   (K = 6, scalar FMA is optimal)
// ---------------------------------------------------------------------------
__global__ void k_gemm_in(const float* __restrict__ X, const float* __restrict__ W1,
                          float* __restrict__ H) {
    long long idx = (long long)blockIdx.x * blockDim.x + threadIdx.x;
    if (idx >= (long long)NNODES * HID) return;
    int i = (int)(idx >> 6);
    int c = (int)(idx & 63);
    const float* f = X + (long long)i * INDIM;
    float s = 0.f;
#pragma unroll
    for (int k = 0; k < INDIM; ++k) s = fmaf(f[k], W1[k * HID + c], s);
    H[idx] = s;
}

// ---------------------------------------------------------------------------
// GCN layer 2 transform: H = X @ W  (N x 64 @ 64 x 64), WMMA, grid-stride tiles,
// B fragments in registers (zero LDS traffic in steady state).
// ---------------------------------------------------------------------------
__global__ void __launch_bounds__(256) k_gemm64(const float* __restrict__ X,
                                                const float* __restrict__ W,
                                                float* __restrict__ H) {
    __shared__ __align__(32) _Float16 Wswz[8192];
    fill_wswz(Wswz, W, threadIdx.x, 256);
    __syncthreads();

    const int lane = threadIdx.x & 31;
    const int sel = lane >> 4, col = lane & 15, row = lane & 15;

    v16h bf[8];
    load_bfrags(Wswz, bf, sel, col);

    const int waveId = blockIdx.x * 8 + (threadIdx.x >> 5);
    const int nWaves = gridDim.x * 8;
    const int nTiles = NNODES / 16;

    for (int tile = waveId; tile < nTiles; tile += nWaves) {
        const float* Arows = X + (long long)tile * 16 * HID;
        v8f acc[4];
#pragma unroll
        for (int n = 0; n < 4; ++n)
#pragma unroll
            for (int r = 0; r < 8; ++r) acc[n][r] = 0.f;

        wave_gemm_tile(Arows, bf, acc, row, sel);

        float* Hrows = H + (long long)tile * 16 * HID;
#pragma unroll
        for (int n = 0; n < 4; ++n)
#pragma unroll
            for (int r = 0; r < 8; ++r)
                Hrows[(sel * 8 + r) * HID + n * 16 + col] = acc[n][r];
    }
}

// ---------------------------------------------------------------------------
// Zero an N*HID f32 buffer (graph-replay safe re-init)
// ---------------------------------------------------------------------------
__global__ void k_zero(float4* __restrict__ p, long long n4) {
    long long i = (long long)blockIdx.x * blockDim.x + threadIdx.x;
    if (i < n4) p[i] = make_float4(0.f, 0.f, 0.f, 0.f);
}

// ---------------------------------------------------------------------------
// Edge scatter: agg[dst] += dinv[src]*dinv[dst] * h[src]   (4 channels/thread)
// ---------------------------------------------------------------------------
__global__ void k_scatter(const int* __restrict__ src, const int* __restrict__ dst,
                          const float* __restrict__ dinv, const float* __restrict__ h,
                          float* __restrict__ agg) {
    long long idx = (long long)blockIdx.x * blockDim.x + threadIdx.x;
    if (idx >= (long long)NEDGES * 16) return;
    int e  = (int)(idx >> 4);
    int c4 = (int)(idx & 15) * 4;
    int s = src[e], d = dst[e];
    float w = dinv[s] * dinv[d];
    float4 hv = *reinterpret_cast<const float4*>(h + (long long)s * HID + c4);
    float* out = agg + (long long)d * HID + c4;
    atomicAdd(out + 0, w * hv.x);
    atomicAdd(out + 1, w * hv.y);
    atomicAdd(out + 2, w * hv.z);
    atomicAdd(out + 3, w * hv.w);
}

// ---------------------------------------------------------------------------
// Finish GCN layer: x = relu(agg + dinv^2 * h (self-loop) + b)   (in place)
// ---------------------------------------------------------------------------
__global__ void k_finish(float* __restrict__ agg, const float* __restrict__ h,
                         const float* __restrict__ dinv, const float* __restrict__ b) {
    long long idx = (long long)blockIdx.x * blockDim.x + threadIdx.x;
    if (idx >= (long long)NNODES * HID) return;
    int i = (int)(idx >> 6);
    int c = (int)(idx & 63);
    float di = dinv[i];
    float v = agg[idx] + di * di * h[idx] + b[c];
    agg[idx] = v > 0.f ? v : 0.f;
}

// ---------------------------------------------------------------------------
// Mean pool: g[gid][c] = mean over 128 contiguous nodes
// ---------------------------------------------------------------------------
__global__ void k_pool(const float* __restrict__ x, float* __restrict__ g) {
    int gid = blockIdx.x;
    int c = threadIdx.x;
    const float* base = x + (long long)gid * NODES_PER_GRAPH * HID + c;
    float s = 0.f;
#pragma unroll 4
    for (int n = 0; n < NODES_PER_GRAPH; ++n) s += base[(long long)n * HID];
    g[(long long)gid * HID + c] = s * (1.0f / NODES_PER_GRAPH);
}

// ---------------------------------------------------------------------------
// pc[j] = Ab1[j] + sum_k pooled[k] * Aw1[64+k][j]   (constant actor contribution)
// ---------------------------------------------------------------------------
__global__ void k_pcontrib(const float* __restrict__ pooled, const float* __restrict__ Aw1,
                           const float* __restrict__ Ab1, float* __restrict__ pc) {
    int j = threadIdx.x;
    float s = Ab1[j];
    for (int k = 0; k < HID; ++k) s = fmaf(pooled[k], Aw1[(HID + k) * HID + j], s);
    pc[j] = s;
}

// ---------------------------------------------------------------------------
// Fused actor + critic heads: 128 graphs per block (8 waves x 16 rows).
//   hiddenA = tanh(g @ Aw1_top + pc); score = (hiddenA . Aw2 + Ab2) * 10
//   hiddenC = tanh(g @ Cw1 + Cb1);    value = hiddenC . Cw2 + Cb2
// ---------------------------------------------------------------------------
__global__ void __launch_bounds__(256) k_heads(
    const float* __restrict__ g, const float* __restrict__ Aw1,
    const float* __restrict__ pc, const float* __restrict__ Aw2,
    const float* __restrict__ Ab2, const float* __restrict__ Cw1,
    const float* __restrict__ Cb1, const float* __restrict__ Cw2,
    const float* __restrict__ Cb2, float* __restrict__ scores,
    float* __restrict__ value) {
    __shared__ __align__(32) _Float16 WAs[8192];  // top half of Aw1 (rows 0..63)
    __shared__ __align__(32) _Float16 WCs[8192];
    __shared__ float hid[128 * HID];

    fill_wswz(WAs, Aw1, threadIdx.x, 256);
    fill_wswz(WCs, Cw1, threadIdx.x, 256);
    __syncthreads();

    const int t = threadIdx.x;
    const int wave = t >> 5, lane = t & 31;
    const int sel = lane >> 4, col = lane & 15, row = lane & 15;
    const int row0 = blockIdx.x * 128 + wave * 16;
    const float* Arows = g + (long long)row0 * HID;

    v16h bf[8];
    v8f acc[4];

    // ---- actor: hiddenA -> LDS, row dot with Aw2 ----
    load_bfrags(WAs, bf, sel, col);
#pragma unroll
    for (int n = 0; n < 4; ++n)
#pragma unroll
        for (int r = 0; r < 8; ++r) acc[n][r] = 0.f;
    wave_gemm_tile(Arows, bf, acc, row, sel);

#pragma unroll
    for (int n = 0; n < 4; ++n)
#pragma unroll
        for (int r = 0; r < 8; ++r) {
            int lr = wave * 16 + sel * 8 + r;
            hid[lr * HID + n * 16 + col] = tanhf(acc[n][r] + pc[n * 16 + col]);
        }
    __syncthreads();
    if (t < 128) {
        float s = Ab2[0];
        for (int c = 0; c < HID; ++c) s = fmaf(hid[t * HID + c], Aw2[c], s);
        scores[blockIdx.x * 128 + t] = s * 10.0f;
    }
    __syncthreads();

    // ---- critic: hiddenC -> LDS, row dot with Cw2 ----
    load_bfrags(WCs, bf, sel, col);
#pragma unroll
    for (int n = 0; n < 4; ++n)
#pragma unroll
        for (int r = 0; r < 8; ++r) acc[n][r] = 0.f;
    wave_gemm_tile(Arows, bf, acc, row, sel);

#pragma unroll
    for (int n = 0; n < 4; ++n)
#pragma unroll
        for (int r = 0; r < 8; ++r) {
            int lr = wave * 16 + sel * 8 + r;
            hid[lr * HID + n * 16 + col] = tanhf(acc[n][r] + Cb1[n * 16 + col]);
        }
    __syncthreads();
    if (t < 128) {
        float s = Cb2[0];
        for (int c = 0; c < HID; ++c) s = fmaf(hid[t * HID + c], Cw2[c], s);
        value[blockIdx.x * 128 + t] = s;
    }
}

// ---------------------------------------------------------------------------
// Per-env policy: mask -> softmax -> (faithful quirk) log_softmax of the
// probabilities -> log_prob / entropy. One block per env.
// ---------------------------------------------------------------------------
__global__ void k_policy(const float* __restrict__ scores, const unsigned char* __restrict__ mask,
                         const int* __restrict__ action, void* __restrict__ dout) {
    __shared__ float s[NJOBS];
    __shared__ float lp[NJOBS];
    const int env = blockIdx.x, j = threadIdx.x;

    float v = scores[env * NJOBS + j];
    if (mask[env * NJOBS + j]) v = -INFINITY;  // where(mask, -inf, scores)
    s[j] = v;
    __syncthreads();

    if (j == 0) {
        float m = -INFINITY;
        for (int k = 0; k < NJOBS; ++k) m = fmaxf(m, s[k]);
        float se = 0.f;
        for (int k = 0; k < NJOBS; ++k) se += expf(s[k] - m);
        float inv = 1.0f / se;
        for (int k = 0; k < NJOBS; ++k) s[k] = expf(s[k] - m) * inv;  // p

        // log_softmax over p (reference passes probabilities as logits)
        float m2 = -INFINITY;
        for (int k = 0; k < NJOBS; ++k) m2 = fmaxf(m2, s[k]);
        float se2 = 0.f;
        for (int k = 0; k < NJOBS; ++k) se2 += expf(s[k] - m2);
        float lse = logf(se2) + m2;
        float ent = 0.f;
        for (int k = 0; k < NJOBS; ++k) {
            float l = s[k] - lse;
            lp[k] = l;
            ent -= expf(l) * l;  // softmax(p) == exp(log_softmax(p))
        }
        float* f = (float*)dout;
        ((int*)dout)[env]   = action[env];       // action passthrough (int bits)
        f[NENVS + env]      = lp[action[env]];   // log_prob
        f[2 * NENVS + env]  = ent;               // entropy
    }
}

// ---------------------------------------------------------------------------
extern "C" void kernel_launch(void* const* d_in, const int* in_sizes, int n_in,
                              void* d_out, int out_size, void* d_ws, size_t ws_size,
                              hipStream_t stream) {
    const float* features = (const float*)d_in[0];
    const int*   edge_idx = (const int*)d_in[1];
    // d_in[2] = batch (contiguous equal-sized graphs; derived analytically)
    const unsigned char* mask = (const unsigned char*)d_in[3];
    const int*   action  = (const int*)d_in[4];
    const float* W1  = (const float*)d_in[5];
    const float* b1  = (const float*)d_in[6];
    const float* W2  = (const float*)d_in[7];
    const float* b2  = (const float*)d_in[8];
    const float* Aw1 = (const float*)d_in[9];
    const float* Ab1 = (const float*)d_in[10];
    const float* Aw2 = (const float*)d_in[11];
    const float* Ab2 = (const float*)d_in[12];
    const float* Cw1 = (const float*)d_in[13];
    const float* Cb1 = (const float*)d_in[14];
    const float* Cw2 = (const float*)d_in[15];
    const float* Cb2 = (const float*)d_in[16];
    const float* pooled = (const float*)d_in[17];

    const int* src = edge_idx;            // edge_index[0]
    const int* dst = edge_idx + NEDGES;   // edge_index[1]

    // Workspace layout (floats)
    float* ws    = (float*)d_ws;
    float* dinv  = ws;                                  // NNODES
    float* bufA  = dinv + NNODES;                       // NNODES*HID (h)
    float* bufB  = bufA + (long long)NNODES * HID;      // NNODES*HID (agg/x)
    float* gbuf  = bufB + (long long)NNODES * HID;      // NGRAPHS*HID
    float* pc    = gbuf + (long long)NGRAPHS * HID;     // HID
    float* score = pc + HID;                            // NGRAPHS

    float* out_value = (float*)d_out + 3 * NENVS;       // [NGRAPHS]

    const long long nelem = (long long)NNODES * HID;
    const int B = 256;

    // degree + symmetric normalization
    k_init_deg<<<(NNODES + B - 1) / B, B, 0, stream>>>(dinv);
    k_deg_accum<<<(NEDGES + B - 1) / B, B, 0, stream>>>(dst, dinv);
    k_rsqrt_inplace<<<(NNODES + B - 1) / B, B, 0, stream>>>(dinv);

    // ---- GCN layer 1 ----
    k_gemm_in<<<(int)((nelem + B - 1) / B), B, 0, stream>>>(features, W1, bufA);
    k_zero<<<(int)((nelem / 4 + B - 1) / B), B, 0, stream>>>((float4*)bufB, nelem / 4);
    k_scatter<<<(int)(((long long)NEDGES * 16 + B - 1) / B), B, 0, stream>>>(src, dst, dinv, bufA, bufB);
    k_finish<<<(int)((nelem + B - 1) / B), B, 0, stream>>>(bufB, bufA, dinv, b1);

    // ---- GCN layer 2 (WMMA transform, grid-stride) ----
    k_gemm64<<<1024, 256, 0, stream>>>(bufB, W2, bufA);
    k_zero<<<(int)((nelem / 4 + B - 1) / B), B, 0, stream>>>((float4*)bufB, nelem / 4);
    k_scatter<<<(int)(((long long)NEDGES * 16 + B - 1) / B), B, 0, stream>>>(src, dst, dinv, bufA, bufB);
    k_finish<<<(int)((nelem + B - 1) / B), B, 0, stream>>>(bufB, bufA, dinv, b2);

    // ---- pool + heads ----
    k_pool<<<NGRAPHS, HID, 0, stream>>>(bufB, gbuf);
    k_pcontrib<<<1, HID, 0, stream>>>(pooled, Aw1, Ab1, pc);
    k_heads<<<NGRAPHS / 128, 256, 0, stream>>>(gbuf, Aw1, pc, Aw2, Ab2,
                                               Cw1, Cb1, Cw2, Cb2, score, out_value);

    // ---- policy distribution ----
    k_policy<<<NENVS, NJOBS, 0, stream>>>(score, mask, action, d_out);
}
